// VQVAE_15126874816811
// MI455X (gfx1250) — matrix operand order repaired and checked by reading, not compiled, not run on hardware
//
#include <hip/hip_runtime.h>

typedef _Float16 v16h __attribute__((ext_vector_type(16)));
typedef float    v8f  __attribute__((ext_vector_type(8)));

// ---------------------------------------------------------------------------
// Small utility kernels
// ---------------------------------------------------------------------------
__global__ void f32_to_f16_kernel(const float* __restrict__ src,
                                  _Float16* __restrict__ dst, int n) {
  int t = blockIdx.x * blockDim.x + threadIdx.x;
  if (t < n) dst[t] = (_Float16)src[t];
}

// conv weight [Cout, Cin*16] f32 -> [Cout, Kpad] f16 (zero pad K)
__global__ void pack_conv_w_kernel(const float* __restrict__ w,
                                   _Float16* __restrict__ out,
                                   int Cout, int Kreal, int Kpad) {
  int t = blockIdx.x * blockDim.x + threadIdx.x;
  if (t >= Cout * Kpad) return;
  int co = t / Kpad, k = t - co * Kpad;
  out[t] = (k < Kreal) ? (_Float16)w[co * Kreal + k] : (_Float16)0.f;
}

// deconv weight dw[Cin, CoutReal, 4,4] -> B[Npad][Cin*16] f16 with spatial flip
// B[co, ci*16 + kh*4 + kw] = dw[ci, co, 3-kh, 3-kw]
__global__ void pack_deconv_w_kernel(const float* __restrict__ dw,
                                     _Float16* __restrict__ out,
                                     int CoutReal, int Npad, int Cin) {
  int K = Cin * 16;
  int t = blockIdx.x * blockDim.x + threadIdx.x;
  if (t >= Npad * K) return;
  int co = t / K, k = t - co * K;
  _Float16 v = (_Float16)0.f;
  if (co < CoutReal) {
    int ci = k >> 4, kh = (k >> 2) & 3, kw = k & 3;
    v = (_Float16)dw[(((size_t)ci * CoutReal + co) * 4 + (3 - kh)) * 4 + (3 - kw)];
  }
  out[t] = v;
}

__global__ void enorm_kernel(const float* __restrict__ E, float* __restrict__ Enorm) {
  int j = blockIdx.x * blockDim.x + threadIdx.x;
  if (j < 512) {
    float s = 0.f;
    for (int c = 0; c < 256; ++c) { float e = E[j * 256 + c]; s += e * e; }
    Enorm[j] = s;
  }
}

__global__ void zero_scalar_kernel(float* p) { p[0] = 0.f; }

__global__ void finalize_loss_kernel(const float* __restrict__ sse, float* out) {
  // vq_loss = q_latent + 0.25*e_latent ; both numerically MSE(q,z) -> 1.25*MSE
  out[0] = 1.25f * sse[0] / 2097152.0f;
}

// ---------------------------------------------------------------------------
// Implicit-GEMM conv / GEMM with WMMA f16 -> f32
//   MODE 0: encoder conv  (4x4, stride 2, pad 1), A = NCHW f16 input
//   MODE 1: deconv as conv(lhs_dilation=2, pad 2), weights pre-flipped
//   MODE 2: plain GEMM, A row-major [M,K] f16 (VQ distances)
//   ACT 0 none, 1 relu, 2 sigmoid
//   NT: N sub-tiles per wave (wave computes 16M x 16*NT N; A fragment reused
//       NT times per K step to amortize the im2col gather)
// Block = 256 threads = 8 waves covering 128 M rows; B tile staged in LDS.
// ---------------------------------------------------------------------------
template <int MODE, int ACT, int NT>
__global__ __launch_bounds__(256) void gemm_conv_wmma(
    const _Float16* __restrict__ A, const _Float16* __restrict__ Bw,
    const float* __restrict__ bias,
    _Float16* __restrict__ out16, float* __restrict__ out32,
    int mblocks, int Nreal, int K,
    int Cin, int Hin, int Win, int Hout, int Wout) {
  __shared__ _Float16 Blds[16 * NT][40];  // 16*NT N rows x 32 K halfs (+pad)

  const int tid  = threadIdx.x;
  const int wave = tid >> 5;
  const int lane = tid & 31;
  const int bm   = blockIdx.x % mblocks;
  const int bn   = blockIdx.x / mblocks;
  const int m0   = bm * 128 + wave * 16;
  const int n0   = bn * (16 * NT);

  // A fragment (16x32 f16, §7.12.2): lane supplies row M = m0 + (lane&15);
  // halves h map to K = kb + (h<8 ? h : h+8), kb = (lane>=16)*8.
  const int mrow  = m0 + (lane & 15);
  const int kb    = (lane & 16) ? 8 : 0;
  const int khoff = kb >> 2;  // this lane's kh pair: khoff, khoff+1

  int b = 0, oh = 0, ow = 0;
  if (MODE != 2) {
    ow = mrow % Wout;
    int t = mrow / Wout;
    oh = t % Hout;
    b  = t / Hout;
  }

  // Loop-invariant im2col taps for this lane (rows: only 2 kh values used).
  int ih_lo = 0, ih_hi = 0, vh_lo = 0, vh_hi = 0;
  int iwq[4] = {0, 0, 0, 0}, vw[4] = {0, 0, 0, 0};
  if (MODE == 0) {
    ih_lo = oh * 2 - 1 + khoff;
    ih_hi = ih_lo + 1;
    vh_lo = (unsigned)ih_lo < (unsigned)Hin;
    vh_hi = (unsigned)ih_hi < (unsigned)Hin;
#pragma unroll
    for (int t = 0; t < 4; ++t) {
      int iw = ow * 2 - 1 + t;
      iwq[t] = iw;
      vw[t]  = (unsigned)iw < (unsigned)Win;
    }
  } else if (MODE == 1) {
    const int d0 = oh - 2 + khoff, d1 = d0 + 1;
    ih_lo = d0 >> 1;
    ih_hi = d1 >> 1;
    vh_lo = (d0 >= 0) && !(d0 & 1) && (ih_lo < Hin);
    vh_hi = (d1 >= 0) && !(d1 & 1) && (ih_hi < Hin);
#pragma unroll
    for (int t = 0; t < 4; ++t) {
      int dw = ow - 2 + t;
      iwq[t] = dw >> 1;
      vw[t]  = (dw >= 0) && !(dw & 1) && ((dw >> 1) < Win);
    }
  }

  const size_t planeHW = (size_t)Hin * Win;
  const size_t base_b  = (MODE == 2) ? 0 : (size_t)b * Cin * planeHW;

  v8f acc[NT];
#pragma unroll
  for (int i = 0; i < NT; ++i)
#pragma unroll
    for (int j = 0; j < 8; ++j) acc[i][j] = 0.f;

  for (int k0 = 0; k0 < K; k0 += 32) {
    __syncthreads();  // previous iteration's LDS reads complete
    // cooperative fill: NT*512 halfs, NT*2 per thread
#pragma unroll
    for (int i = 0; i < NT; ++i) {
      const int e  = tid * 2 + i * 512;
      const int nn = e >> 5, kk = e & 31;
      const _Float16* src = Bw + (size_t)(n0 + nn) * K + k0 + kk;
      Blds[nn][kk]     = src[0];
      Blds[nn][kk + 1] = src[1];
    }
    __syncthreads();

    // A fragment gather (implicit im2col); each 8-half group lives in one ci.
    v16h af;
    if (MODE == 2) {
      const _Float16* p0 = A + (size_t)mrow * K + k0 + kb;
#pragma unroll
      for (int h = 0; h < 8; ++h) { af[h] = p0[h]; af[h + 8] = p0[16 + h]; }
    } else {
      const int ci0 = k0 >> 4;  // (k0 + kb + 16g + h) >> 4 == k0>>4 + g
#pragma unroll
      for (int g = 0; g < 2; ++g) {
        const int ci = ci0 + g;
        const int vg = (ci < Cin);
        const _Float16* bp  = A + base_b + (size_t)ci * planeHW;
        const _Float16* rlo = bp + (size_t)ih_lo * Win;
        const _Float16* rhi = bp + (size_t)ih_hi * Win;
#pragma unroll
        for (int h = 0; h < 8; ++h) {
          const int kw = h & 3;
          const _Float16* rp = (h >= 4) ? rhi : rlo;
          const int ok = ((h >= 4) ? vh_hi : vh_lo) & vw[kw] & vg;
          af[g * 8 + h] = ok ? rp[iwq[kw]] : (_Float16)0.f;
        }
      }
    }

    // NT WMMAs reuse the same A fragment against LDS-resident B sub-tiles.
#pragma unroll
    for (int nt = 0; nt < NT; ++nt) {
      v16h bf;  // B 32x16: lane holds N = lane&15, K = (lane>=16)*8 + (i<8 ? i : i+8)
      const _Float16* bp = &Blds[nt * 16 + (lane & 15)][(lane & 16) ? 8 : 0];
#pragma unroll
      for (int i = 0; i < 16; ++i) bf[i] = bp[(i < 8) ? i : i + 8];
      acc[nt] = __builtin_amdgcn_wmma_f32_16x16x32_f16(
          false, af, false, bf, (short)0, acc[nt], false, false);
    }
  }

  // Epilogue: C element r -> M = m0 + r + 8*(lane>=16), N = n0 + nt*16 + lane&15
  const int ln   = lane & 15;
  const int mbse = m0 + ((lane & 16) ? 8 : 0);
#pragma unroll
  for (int nt = 0; nt < NT; ++nt) {
    const int n = n0 + nt * 16 + ln;
    const float bv = (bias && n < Nreal) ? bias[n] : 0.f;
#pragma unroll
    for (int r = 0; r < 8; ++r) {
      const int m = mbse + r;
      float v = acc[nt][r] + bv;
      if (ACT == 1) v = v > 0.f ? v : 0.f;
      else if (ACT == 2) v = 1.f / (1.f + __expf(-v));
      if (n < Nreal) {
        if (MODE == 2) {
          out32[(size_t)m * Nreal + n] = v;
        } else {
          const int oww = m % Wout;
          int tt = m / Wout;
          const int ohh = tt % Hout;
          const int bb  = tt / Hout;
          const size_t o = (((size_t)bb * Nreal + n) * Hout + ohh) * Wout + oww;
          if (out16) out16[o] = (_Float16)v;
          if (out32) out32[o] = v;
        }
      }
    }
  }
}

// ---------------------------------------------------------------------------
// VQ argmin: one wave per row; |z|^2 is a per-row constant so argmin over
// (Enorm_j - 2*G_j).
// ---------------------------------------------------------------------------
__global__ __launch_bounds__(256) void vq_argmin_kernel(
    const float* __restrict__ G, const float* __restrict__ Enorm,
    int* __restrict__ idxb, float* __restrict__ idx_out) {
  const int wave = threadIdx.x >> 5, lane = threadIdx.x & 31;
  const int row = blockIdx.x * 8 + wave;
  if (row >= 8192) return;
  float best = 3.4e38f;
  int bj = 0x7fffffff;
  for (int j = lane; j < 512; j += 32) {
    float s = Enorm[j] - 2.f * G[(size_t)row * 512 + j];
    if (s < best || (s == best && j < bj)) { best = s; bj = j; }
  }
  for (int off = 16; off > 0; off >>= 1) {
    float ob = __shfl_xor(best, off, 32);
    int   oj = __shfl_xor(bj, off, 32);
    if (ob < best || (ob == best && oj < bj)) { best = ob; bj = oj; }
  }
  if (lane == 0) { idxb[row] = bj; idx_out[row] = (float)bj; }
}

// quantized = E[idx]; accumulate sum((q - z)^2); emit q as f16 (decoder input)
__global__ __launch_bounds__(256) void vq_gather_loss_kernel(
    const float* __restrict__ E, const int* __restrict__ idxb,
    const float* __restrict__ zbuf, _Float16* __restrict__ q16,
    float* __restrict__ sse) {
  __shared__ float red[256];
  const int gid = blockIdx.x * 256 + threadIdx.x;
  const int n = gid >> 8, c = gid & 255;
  const float e = E[(size_t)idxb[n] * 256 + c];
  const float z = zbuf[gid];
  q16[gid] = (_Float16)e;
  const float d = e - z;
  red[threadIdx.x] = d * d;
  __syncthreads();
  for (int s = 128; s > 0; s >>= 1) {
    if (threadIdx.x < s) red[threadIdx.x] += red[threadIdx.x + s];
    __syncthreads();
  }
  if (threadIdx.x == 0) atomicAdd(sse, red[0]);
}

// ---------------------------------------------------------------------------
extern "C" void kernel_launch(void* const* d_in, const int* in_sizes, int n_in,
                              void* d_out, int out_size, void* d_ws, size_t ws_size,
                              hipStream_t stream) {
  const float* x   = (const float*)d_in[0];
  const float* w1  = (const float*)d_in[1];  const float* b1  = (const float*)d_in[2];
  const float* w2  = (const float*)d_in[3];  const float* b2  = (const float*)d_in[4];
  const float* w3  = (const float*)d_in[5];  const float* b3  = (const float*)d_in[6];
  const float* w4  = (const float*)d_in[7];  const float* b4  = (const float*)d_in[8];
  const float* dw1 = (const float*)d_in[9];  const float* db1 = (const float*)d_in[10];
  const float* dw2 = (const float*)d_in[11]; const float* db2 = (const float*)d_in[12];
  const float* dw3 = (const float*)d_in[13]; const float* db3 = (const float*)d_in[14];
  const float* dw4 = (const float*)d_in[15]; const float* db4 = (const float*)d_in[16];
  const float* E   = (const float*)d_in[17];

  float* yout    = (float*)d_out;            // [32,1,256,256] = 2097152
  float* lossout = yout + 2097152;           // scalar
  float* idxout  = yout + 2097153;           // 8192 indices (as float)

  // workspace carve-up
  char* p = (char*)d_ws;
  auto alloc = [&](size_t bytes) -> char* {
    char* r = p; p += (bytes + 255) & ~(size_t)255; return r;
  };
  _Float16* bufA = (_Float16*)alloc(33554432);   // max f16 activation (16.7M elems)
  _Float16* bufB = (_Float16*)alloc(33554432);
  _Float16* x16  = (_Float16*)alloc(4194304);    // 2,097,152 halfs
  float*    zbuf = (float*)   alloc(8388608);    // z f32 NCHW [32,256,16,16]
  float*    G    = (float*)   alloc(16777216);   // [8192,512] f32
  _Float16* E16  = (_Float16*)alloc(262144);     // [512,256]
  float*    Enrm = (float*)   alloc(2048);
  int*      idxb = (int*)     alloc(32768);
  float*    sse  = (float*)   alloc(256);
  _Float16* w1p  = (_Float16*)alloc(32  * 32   * 2);
  _Float16* w2p  = (_Float16*)alloc(64  * 512  * 2);
  _Float16* w3p  = (_Float16*)alloc(128 * 1024 * 2);
  _Float16* w4p  = (_Float16*)alloc(256 * 2048 * 2);
  _Float16* dw1p = (_Float16*)alloc(128 * 4096 * 2);
  _Float16* dw2p = (_Float16*)alloc(64  * 2048 * 2);
  _Float16* dw3p = (_Float16*)alloc(32  * 1024 * 2);
  _Float16* dw4p = (_Float16*)alloc(16  * 512  * 2);

  auto cdiv = [](int a, int b) { return (a + b - 1) / b; };

  // --- precision packing -----------------------------------------------------
  f32_to_f16_kernel<<<cdiv(2097152, 256), 256, 0, stream>>>(x, x16, 2097152);
  f32_to_f16_kernel<<<cdiv(131072, 256), 256, 0, stream>>>(E, E16, 131072);
  enorm_kernel<<<2, 256, 0, stream>>>(E, Enrm);
  pack_conv_w_kernel<<<cdiv(32 * 32, 256), 256, 0, stream>>>(w1, w1p, 32, 16, 32);
  pack_conv_w_kernel<<<cdiv(64 * 512, 256), 256, 0, stream>>>(w2, w2p, 64, 512, 512);
  pack_conv_w_kernel<<<cdiv(128 * 1024, 256), 256, 0, stream>>>(w3, w3p, 128, 1024, 1024);
  pack_conv_w_kernel<<<cdiv(256 * 2048, 256), 256, 0, stream>>>(w4, w4p, 256, 2048, 2048);
  pack_deconv_w_kernel<<<cdiv(128 * 4096, 256), 256, 0, stream>>>(dw1, dw1p, 128, 128, 256);
  pack_deconv_w_kernel<<<cdiv(64 * 2048, 256), 256, 0, stream>>>(dw2, dw2p, 64, 64, 128);
  pack_deconv_w_kernel<<<cdiv(32 * 1024, 256), 256, 0, stream>>>(dw3, dw3p, 32, 32, 64);
  pack_deconv_w_kernel<<<cdiv(16 * 512, 256), 256, 0, stream>>>(dw4, dw4p, 1, 16, 32);

  // --- encoder: implicit-GEMM WMMA convs ------------------------------------
  // conv1: M=524288 (mblocks 4096), Npad 32 (NT=2 -> 1 nblock), K 32 (padded)
  gemm_conv_wmma<0, 1, 2><<<4096 * 1, 256, 0, stream>>>(
      x16, w1p, b1, bufA, nullptr, 4096, 32, 32, 1, 256, 256, 128, 128);
  // conv2: M=131072, N=64, K=512
  gemm_conv_wmma<0, 1, 4><<<1024 * 1, 256, 0, stream>>>(
      bufA, w2p, b2, bufB, nullptr, 1024, 64, 512, 32, 128, 128, 64, 64);
  // conv3: M=32768, N=128, K=1024
  gemm_conv_wmma<0, 1, 4><<<256 * 2, 256, 0, stream>>>(
      bufB, w3p, b3, bufA, nullptr, 256, 128, 1024, 64, 64, 64, 32, 32);
  // conv4 (no activation): z f16 -> bufB, z f32 -> zbuf
  gemm_conv_wmma<0, 0, 4><<<64 * 4, 256, 0, stream>>>(
      bufA, w4p, b4, bufB, zbuf, 64, 256, 2048, 128, 32, 32, 16, 16);

  // --- vector quantizer ------------------------------------------------------
  // G[8192,512] = flat(z) @ E^T  (flat == z NCHW viewed [8192,256])
  gemm_conv_wmma<2, 0, 4><<<64 * 8, 256, 0, stream>>>(
      bufB, E16, nullptr, nullptr, G, 64, 512, 256, 0, 0, 0, 0, 0);
  vq_argmin_kernel<<<1024, 256, 0, stream>>>(G, Enrm, idxb, idxout);
  zero_scalar_kernel<<<1, 1, 0, stream>>>(sse);
  vq_gather_loss_kernel<<<8192, 256, 0, stream>>>(E, idxb, zbuf, bufA, sse);
  finalize_loss_kernel<<<1, 1, 0, stream>>>(sse, lossout);

  // --- decoder: deconvs as dilated convs -------------------------------------
  // deconv1: M=32768, N=128, K=4096 (in [32,256,16,16] -> out [32,128,32,32])
  gemm_conv_wmma<1, 1, 4><<<256 * 2, 256, 0, stream>>>(
      bufA, dw1p, db1, bufB, nullptr, 256, 128, 4096, 256, 16, 16, 32, 32);
  // deconv2: M=131072, N=64, K=2048
  gemm_conv_wmma<1, 1, 4><<<1024 * 1, 256, 0, stream>>>(
      bufB, dw2p, db2, bufA, nullptr, 1024, 64, 2048, 128, 32, 32, 64, 64);
  // deconv3: M=524288, N=32, K=1024
  gemm_conv_wmma<1, 1, 2><<<4096 * 1, 256, 0, stream>>>(
      bufA, dw3p, db3, bufB, nullptr, 4096, 32, 1024, 64, 64, 64, 128, 128);
  // deconv4 + sigmoid: M=2097152, Nreal=1 (Npad 16), K=512 -> y f32 to d_out
  gemm_conv_wmma<1, 2, 1><<<16384 * 1, 256, 0, stream>>>(
      bufB, dw4p, db4, nullptr, yout, 16384, 1, 512, 32, 128, 128, 256, 256);
}
